// MultiHeadAttention_54735063220568
// MI455X (gfx1250) — matrix operand order repaired and checked by reading, not compiled, not run on hardware
//
#include <hip/hip_runtime.h>

// ---------------------------------------------------------------------------
// MI455X (gfx1250) multi-head attention, bf16 WMMA pipeline.
//   B=2, T=2048, E=1024, H=16, D=64.
// The raw .view(B,H,T,D) reshape makes each head a contiguous (T x D) slab of
// the flat (B*T,E) projection buffer -> plain row-major per-head tensors.
// Attention computes S^T = K*Q^T and O^T = V^T*P^T so softmax state is one
// scalar per lane. 32 queries per wave. K/V tiles are staged into LDS once
// per workgroup via the Tensor Data Mover (double-buffered), instead of each
// wave re-reading them from L2.
// ---------------------------------------------------------------------------

typedef __bf16 bf16_t;
typedef __bf16 v16bf __attribute__((ext_vector_type(16)));
typedef float  v8f   __attribute__((ext_vector_type(8)));
typedef unsigned int u32x4 __attribute__((ext_vector_type(4)));
typedef unsigned int tdm_u4 __attribute__((ext_vector_type(4)));
typedef int tdm_i8 __attribute__((ext_vector_type(8)));
typedef int tdm_i4 __attribute__((ext_vector_type(4)));

#if __has_builtin(__builtin_amdgcn_tensor_load_to_lds)
#define USE_TDM 1
#else
#define USE_TDM 0
#endif

union FragU { v16bf v; u32x4 u[2]; };

__device__ __forceinline__ void sched_fence() {
#if __has_builtin(__builtin_amdgcn_sched_barrier)
  __builtin_amdgcn_sched_barrier(0);
#endif
}

__device__ __forceinline__ void wait_tensorcnt0() {
#if __has_builtin(__builtin_amdgcn_s_wait_tensorcnt)
  __builtin_amdgcn_s_wait_tensorcnt(0);
#else
  asm volatile("s_wait_tensorcnt 0" ::: "memory");
#endif
}

// ---- TDM 2-D tile load (ISA 8.3-8.5): global -> LDS with row padding -------
// data_size=2B; pad_interval/pad_amount give the LDS pitch the WMMA fragment
// loaders want. Issued wave-uniform (EXEC is ignored by TENSOR ops).
__device__ __forceinline__ void tdm_load_tile_2d(
    unsigned lds_byte_off, const bf16_t* gptr,
    unsigned tensor_d0, unsigned tensor_d1,
    unsigned tile_d0, unsigned tile_d1, unsigned row_stride_elems,
    unsigned pad_interval_code, unsigned pad_amount_code) {
#if USE_TDM
  unsigned long long ga = (unsigned long long)gptr;
  tdm_u4 g0;
  g0[0] = 1u;  // count=1, user descriptor, gather off
  g0[1] = lds_byte_off;
  g0[2] = (unsigned)ga;
  g0[3] = (unsigned)((ga >> 32) & 0x01FFFFFFu) | (2u << 30);  // type=2 (image)
  tdm_i8 g1;
  g1[0] = (int)((1u << 16) | (1u << 20) |               // data_size=2B, pad_en
                (pad_interval_code << 22) | (pad_amount_code << 25));
  g1[1] = (int)(tensor_d0 << 16);                       // tensor_dim0 [79:48]
  g1[2] = (int)((tensor_d0 >> 16) | (tensor_d1 << 16)); // tensor_dim1 [111:80]
  g1[3] = (int)((tensor_d1 >> 16) | (tile_d0 << 16));   // tile_dim0 [127:112]
  g1[4] = (int)tile_d1;                                 // tile_dim1 [143:128]
  g1[5] = (int)row_stride_elems;                        // dim0_stride [207:160]
  g1[6] = 0;
  g1[7] = 0;
  tdm_i4 z4 = {0, 0, 0, 0};
#if __clang_major__ >= 23
  tdm_i8 z8 = {0, 0, 0, 0, 0, 0, 0, 0};
  __builtin_amdgcn_tensor_load_to_lds(g0, g1, z4, z4, z8, 0);
#else
  __builtin_amdgcn_tensor_load_to_lds(g0, g1, z4, z4, 0);
#endif
#else
  (void)lds_byte_off; (void)gptr; (void)tensor_d0; (void)tensor_d1;
  (void)tile_d0; (void)tile_d1; (void)row_stride_elems;
  (void)pad_interval_code; (void)pad_amount_code;
#endif
}

// ---- WMMA fragment loaders (work on global or LDS pointers) ----------------
// A-operand 16x32 bf16: lanes 0-15 row M=lane, K={0..7,16..23}; lanes 16-31
// K={8..15,24..31}.
__device__ __forceinline__ v16bf load_frag_a(const bf16_t* p, int ld, int row0, int k0) {
  const int lane = threadIdx.x & 31;
  const bf16_t* base = p + (size_t)(row0 + (lane & 15)) * ld + (k0 + ((lane >> 4) << 3));
  FragU f;
  f.u[0] = *(const u32x4*)(base);
  f.u[1] = *(const u32x4*)(base + 16);
  return f.v;
}

// B-operand 32x16 bf16: lanes 0-15 column N=lane, K={0..7,16..23}; lanes
// 16-31 K={8..15,24..31}. `p` is column-major source: p[col*ld + k].
__device__ __forceinline__ v16bf load_frag_b(const bf16_t* p, int ld, int col0, int k0) {
  const int lane = threadIdx.x & 31;
  const bf16_t* base = p + (size_t)(col0 + (lane & 15)) * ld + (k0 + ((lane >> 4) << 3));
  FragU f;
  f.u[0] = *(const u32x4*)(base);
  f.u[1] = *(const u32x4*)(base + 16);
  return f.v;
}

__device__ __forceinline__ v8f wmma_bf16(v16bf a, v16bf b, v8f c) {
  return __builtin_amdgcn_wmma_f32_16x16x32_bf16(false, a, false, b, (short)0, c,
                                                 false, false);
}

__device__ __forceinline__ unsigned pack_bf16x2(float a, float b) {
  union { bf16_t h[2]; unsigned u; } t;
  t.h[0] = (bf16_t)a;
  t.h[1] = (bf16_t)b;
  return t.u;
}

// ---- prep kernels ----------------------------------------------------------
__global__ void cvt_f32_to_bf16(const float* __restrict__ src, bf16_t* __restrict__ dst, int n) {
  int i = (blockIdx.x * blockDim.x + threadIdx.x) * 4;
  if (i >= n) return;
  float4 v = *(const float4*)(src + i);
  dst[i + 0] = (bf16_t)v.x;
  dst[i + 1] = (bf16_t)v.y;
  dst[i + 2] = (bf16_t)v.z;
  dst[i + 3] = (bf16_t)v.w;
}

__global__ void transpose_cvt(const float* __restrict__ W, bf16_t* __restrict__ WT,
                              int K, int N) {
  int idx = blockIdx.x * blockDim.x + threadIdx.x;
  if (idx >= K * N) return;
  int k = idx / N, n = idx - k * N;
  WT[(size_t)n * K + k] = (bf16_t)W[idx];
}

// ---- GEMM: C(MxN) = A(MxK,bf16) * B(KxN), B given as Bt(NxK,bf16) ----------
// MODE 0: bf16 row-major; MODE 1: bf16 transposed per head (Vt); MODE 2: f32.
template <int MODE>
__global__ void gemm_bf16_kernel(const bf16_t* __restrict__ A, const bf16_t* __restrict__ Bt,
                                 void* __restrict__ Cout, int M, int N, int K) {
  const int wave = threadIdx.x >> 5;
  const int lane = threadIdx.x & 31;
  const int tilesN = N >> 6;
  const int wt = blockIdx.x * (blockDim.x >> 5) + wave;
  const int mt = wt / tilesN, nt = wt - mt * tilesN;
  const int m0 = mt * 32, n0 = nt * 64;
  if (m0 >= M) return;

  v8f acc[2][4] = {};
  for (int k0 = 0; k0 < K; k0 += 32) {
    v16bf a0 = load_frag_a(A, K, m0, k0);
    v16bf a1 = load_frag_a(A, K, m0 + 16, k0);
#pragma unroll
    for (int s = 0; s < 4; ++s) {
      v16bf b = load_frag_b(Bt, K, n0 + s * 16, k0);
      acc[0][s] = wmma_bf16(a0, b, acc[0][s]);
      acc[1][s] = wmma_bf16(a1, b, acc[1][s]);
    }
  }

  const int colLane = lane & 15;
  const int rowHi = (lane >> 4) << 3;
#pragma unroll
  for (int h = 0; h < 2; ++h) {
#pragma unroll
    for (int s = 0; s < 4; ++s) {
#pragma unroll
      for (int j = 0; j < 8; ++j) {
        int m = m0 + h * 16 + rowHi + j;
        int n = n0 + s * 16 + colLane;
        float val = acc[h][s][j];
        if (MODE == 0) {
          ((bf16_t*)Cout)[(size_t)m * N + n] = (bf16_t)val;
        } else if (MODE == 2) {
          ((float*)Cout)[(size_t)m * N + n] = val;
        } else {
          int bh = m >> 7;
          int d = n & 63;
          int t = ((m & 127) << 4) | (n >> 6);
          ((bf16_t*)Cout)[((size_t)bh * 64 + d) * 2048 + t] = (bf16_t)val;
        }
      }
    }
  }
}

// ---- flash attention: TDM-staged K/V, 32 queries / wave --------------------
// grid = (T/256, B*H), block = 256 (8 waves).
__global__ __launch_bounds__(256, 1)
void attn_kernel(const bf16_t* __restrict__ Q, const bf16_t* __restrict__ Kh,
                 const bf16_t* __restrict__ Vt, bf16_t* __restrict__ O) {
  constexpr int T = 2048, D = 64;
  constexpr int KP = 72;   // K tile LDS pitch (64 + 8): conflict-free, 16B-aligned
  constexpr int VP = 40;   // V tile LDS pitch (32 + 8)
  constexpr int LDP = 40;  // P^T staging pitch
  struct Shared {
    bf16_t k[2][32 * KP];      // 2 x 4608 B
    bf16_t v[2][64 * VP];      // 2 x 5120 B
    bf16_t p[8][2][16 * LDP];  // 20480 B
  };
  __shared__ __align__(16) Shared sm;  // single LDS object -> base offset 0

  const int wave = threadIdx.x >> 5;
  const int lane = threadIdx.x & 31;
  const int tid = threadIdx.x;
  const int bh = blockIdx.y;
  const int t0 = (blockIdx.x * 8 + wave) * 32;

  const bf16_t* Qp = Q + (size_t)bh * T * D;
  const bf16_t* Kp = Kh + (size_t)bh * T * D;
  const bf16_t* Vp = Vt + (size_t)bh * D * T;  // (64 x T) = V^T
  bf16_t* Op = O + (size_t)bh * T * D;

  // Q as B-operand (col = query t, k = d); two 16-query groups, loaded once.
  v16bf qb[2][2];
#pragma unroll
  for (int g = 0; g < 2; ++g) {
    qb[g][0] = load_frag_b(Qp, D, t0 + g * 16, 0);
    qb[g][1] = load_frag_b(Qp, D, t0 + g * 16, 32);
  }

  v8f acc[2][4] = {};
  float mrun[2] = {-1e30f, -1e30f};
  float lrun[2] = {0.f, 0.f};
  const int tcol = lane & 15;
  const int hb = lane >> 4;

  const unsigned kOff0 = (unsigned)((char*)&sm.k[0][0] - (char*)&sm);
  const unsigned kOff1 = (unsigned)((char*)&sm.k[1][0] - (char*)&sm);
  const unsigned vOff0 = (unsigned)((char*)&sm.v[0][0] - (char*)&sm);
  const unsigned vOff1 = (unsigned)((char*)&sm.v[1][0] - (char*)&sm);

#if !USE_TDM
  // fallback cooperative-copy mapping (256 threads, 16B each)
  const int kr = tid >> 3, ks = (tid & 7) * 8;   // K: 32 rows x 64
  const int vr = tid >> 2, vs = (tid & 3) * 8;   // V: 64 rows x 32
  u32x4 kreg, vreg;
#endif

  // ---- prologue: stage block 0 into buffer 0 ----
#if USE_TDM
  if (wave == 0) {
    // K tile 32x64 @ pitch 72: 32 dwords/row (code 4) + 4 dwords pad (code 3)
    tdm_load_tile_2d(kOff0, Kp, 64, T, 64, 32, 64, 4, 3);
    // V tile 64x32 @ pitch 40: 16 dwords/row (code 3) + 4 dwords pad (code 3)
    tdm_load_tile_2d(vOff0, Vp, T, 64, 32, 64, T, 3, 3);
    wait_tensorcnt0();
  }
  __syncthreads();
#else
  kreg = *(const u32x4*)(Kp + (size_t)kr * D + ks);
  vreg = *(const u32x4*)(Vp + (size_t)vr * T + vs);
  *(u32x4*)(&sm.k[0][kr * KP + ks]) = kreg;
  *(u32x4*)(&sm.v[0][vr * VP + vs]) = vreg;
  __syncthreads();
#endif

  int par = 0;
  for (int s0 = 0; s0 < T; s0 += 32) {
    const bool haveNext = (s0 + 32 < T);
#if USE_TDM
    if (wave == 0 && haveNext) {
      tdm_load_tile_2d(par ? kOff0 : kOff1, Kp + (size_t)(s0 + 32) * D,
                       64, T, 64, 32, 64, 4, 3);
      tdm_load_tile_2d(par ? vOff0 : vOff1, Vp + (s0 + 32),
                       T, 64, 32, 64, T, 3, 3);
    }
#else
    if (haveNext) {
      kreg = *(const u32x4*)(Kp + (size_t)(s0 + 32 + kr) * D + ks);
      vreg = *(const u32x4*)(Vp + (size_t)vr * T + (s0 + 32) + vs);
    }
#endif
    const bf16_t* kb = &sm.k[par][0];
    const bf16_t* vb = &sm.v[par][0];

    // S^T tiles for both query groups; K frags from LDS, 2 live at a time.
    v8f sa[2], sb[2];
    {
      const v16bf ka0 = load_frag_a(kb, KP, 0, 0);
      const v16bf ka1 = load_frag_a(kb, KP, 0, 32);
      v8f z = {};
      z = wmma_bf16(ka0, qb[0][0], z);
      sa[0] = wmma_bf16(ka1, qb[0][1], z);
      v8f z2 = {};
      z2 = wmma_bf16(ka0, qb[1][0], z2);
      sa[1] = wmma_bf16(ka1, qb[1][1], z2);
    }
    sched_fence();
    {
      const v16bf ka2 = load_frag_a(kb, KP, 16, 0);
      const v16bf ka3 = load_frag_a(kb, KP, 16, 32);
      v8f z = {};
      z = wmma_bf16(ka2, qb[0][0], z);
      sb[0] = wmma_bf16(ka3, qb[0][1], z);
      v8f z2 = {};
      z2 = wmma_bf16(ka2, qb[1][0], z2);
      sb[1] = wmma_bf16(ka3, qb[1][1], z2);
    }
    sched_fence();

    // softmax per group; stage P^T into per-wave LDS region
#pragma unroll
    for (int g = 0; g < 2; ++g) {
      float pa[8], pb[8];
#pragma unroll
      for (int j = 0; j < 8; ++j) {
        pa[j] = sa[g][j] * 0.125f;  // 1/sqrt(64)
        pb[j] = sb[g][j] * 0.125f;
      }
      float c = fmaxf(pa[0], pb[0]);
#pragma unroll
      for (int j = 1; j < 8; ++j) c = fmaxf(c, fmaxf(pa[j], pb[j]));
      c = fmaxf(c, __shfl_xor(c, 16, 32));

      float mnew = fmaxf(mrun[g], c);
      float alpha = __expf(mrun[g] - mnew);
      mrun[g] = mnew;

      float rs = 0.f;
#pragma unroll
      for (int j = 0; j < 8; ++j) {
        pa[j] = __expf(pa[j] - mnew);
        pb[j] = __expf(pb[j] - mnew);
        rs += pa[j] + pb[j];
      }
      rs += __shfl_xor(rs, 16, 32);
      lrun[g] = alpha * lrun[g] + rs;

#pragma unroll
      for (int i = 0; i < 4; ++i)
#pragma unroll
        for (int j = 0; j < 8; ++j) acc[g][i][j] *= alpha;

      bf16_t* myLds = &sm.p[wave][g][0];
      u32x4 w;
      w[0] = pack_bf16x2(pa[0], pa[1]);
      w[1] = pack_bf16x2(pa[2], pa[3]);
      w[2] = pack_bf16x2(pa[4], pa[5]);
      w[3] = pack_bf16x2(pa[6], pa[7]);
      *(u32x4*)(myLds + tcol * LDP + 8 * hb) = w;
      w[0] = pack_bf16x2(pb[0], pb[1]);
      w[1] = pack_bf16x2(pb[2], pb[3]);
      w[2] = pack_bf16x2(pb[4], pb[5]);
      w[3] = pack_bf16x2(pb[6], pb[7]);
      *(u32x4*)(myLds + tcol * LDP + 16 + 8 * hb) = w;
    }
    asm volatile("s_wait_dscnt 0" ::: "memory");  // per-wave LDS RAW fence
    sched_fence();

    v16bf pf0 = load_frag_b(&sm.p[wave][0][0], LDP, 0, 0);
    v16bf pf1 = load_frag_b(&sm.p[wave][1][0], LDP, 0, 0);

    // O^T += V^T(d x s) @ P^T(s x t); V frags from LDS, shared by groups.
#pragma unroll
    for (int i = 0; i < 4; ++i) {
      const v16bf va = load_frag_a(vb, VP, i * 16, 0);
      acc[0][i] = wmma_bf16(va, pf0, acc[0][i]);
      acc[1][i] = wmma_bf16(va, pf1, acc[1][i]);
    }
    sched_fence();

    // publish next K/V buffer
#if USE_TDM
    if (wave == 0 && haveNext) wait_tensorcnt0();
    __syncthreads();
#else
    __syncthreads();
    if (haveNext) {
      *(u32x4*)(&sm.k[par ^ 1][kr * KP + ks]) = kreg;
      *(u32x4*)(&sm.v[par ^ 1][vr * VP + vs]) = vreg;
    }
    __syncthreads();
#endif
    par ^= 1;
  }

  // normalize + packed b128 stores (per lane: fixed t, contiguous d runs)
#pragma unroll
  for (int g = 0; g < 2; ++g) {
    const float linv = 1.f / lrun[g];
    const size_t trow = (size_t)(t0 + g * 16 + tcol) * D;
#pragma unroll
    for (int i = 0; i < 4; ++i) {
      u32x4 w;
      w[0] = pack_bf16x2(acc[g][i][0] * linv, acc[g][i][1] * linv);
      w[1] = pack_bf16x2(acc[g][i][2] * linv, acc[g][i][3] * linv);
      w[2] = pack_bf16x2(acc[g][i][4] * linv, acc[g][i][5] * linv);
      w[3] = pack_bf16x2(acc[g][i][6] * linv, acc[g][i][7] * linv);
      *(u32x4*)(Op + trow + i * 16 + 8 * hb) = w;
    }
  }
}

// ---------------------------------------------------------------------------
extern "C" void kernel_launch(void* const* d_in, const int* in_sizes, int n_in,
                              void* d_out, int out_size, void* d_ws, size_t ws_size,
                              hipStream_t stream) {
  constexpr int Bn = 2, T = 2048, E = 1024;
  constexpr int M = Bn * T;  // 4096
  const float* x  = (const float*)d_in[0];
  const float* Wq = (const float*)d_in[1];
  const float* Wk = (const float*)d_in[2];
  const float* Wv = (const float*)d_in[3];
  const float* Wo = (const float*)d_in[4];

  char* ws = (char*)d_ws;
  constexpr size_t MB = 1024 * 1024;
  bf16_t* Xb  = (bf16_t*)(ws + 0 * MB);
  bf16_t* WqT = (bf16_t*)(ws + 8 * MB);
  bf16_t* WkT = (bf16_t*)(ws + 10 * MB);
  bf16_t* WvT = (bf16_t*)(ws + 12 * MB);
  bf16_t* WoT = (bf16_t*)(ws + 14 * MB);
  bf16_t* Qb  = (bf16_t*)(ws + 16 * MB);
  bf16_t* Kb  = (bf16_t*)(ws + 24 * MB);
  bf16_t* Vtb = (bf16_t*)(ws + 32 * MB);
  bf16_t* Ob  = (bf16_t*)(ws + 40 * MB);

  {
    int n = M * E;
    cvt_f32_to_bf16<<<n / (256 * 4), 256, 0, stream>>>(x, Xb, n);
    int blocks = (E * E + 255) / 256;
    transpose_cvt<<<blocks, 256, 0, stream>>>(Wq, WqT, E, E);
    transpose_cvt<<<blocks, 256, 0, stream>>>(Wk, WkT, E, E);
    transpose_cvt<<<blocks, 256, 0, stream>>>(Wv, WvT, E, E);
    transpose_cvt<<<blocks, 256, 0, stream>>>(Wo, WoT, E, E);
  }
  {
    int waveTiles = (M / 32) * (E / 64);
    int blocks = waveTiles / 8;
    gemm_bf16_kernel<0><<<blocks, 256, 0, stream>>>(Xb, WqT, (void*)Qb, M, E, E);
    gemm_bf16_kernel<0><<<blocks, 256, 0, stream>>>(Xb, WkT, (void*)Kb, M, E, E);
    gemm_bf16_kernel<1><<<blocks, 256, 0, stream>>>(Xb, WvT, (void*)Vtb, M, E, E);
  }
  {
    dim3 grid(T / 256, Bn * 16);
    attn_kernel<<<grid, 256, 0, stream>>>(Qb, Kb, Vtb, Ob);
  }
  {
    int waveTiles = (M / 32) * (E / 64);
    int blocks = waveTiles / 8;
    gemm_bf16_kernel<2><<<blocks, 256, 0, stream>>>(Ob, WoT, d_out, M, E, E);
  }
  (void)in_sizes; (void)n_in; (void)out_size; (void)ws_size;
}
